// Selfattention_64381559767546
// MI455X (gfx1250) — matrix operand order repaired and checked by reading, not compiled, not run on hardware
//
#include <hip/hip_runtime.h>

// MI455X / gfx1250, wave32. bf16x3 split-precision WMMA self-attention:
// every fp32 operand is split hi/lo bf16; each K=32 chunk runs 3x
// V_WMMA_F32_16X16X32_BF16 (ah*bh + ah*bl + al*bh, f32 accumulate) giving
// ~fp32 accuracy at bf16 matrix-pipe rates. q/k/v (bf16, ~40MB) are
// L2-resident; flash-style 2-pass softmax avoids the 134MB energy tensor.
// i-tile widened to 32 rows/workgroup so every V (and K) load is reused
// across two i-subtiles, halving the dominant L2 read stream.

typedef __attribute__((ext_vector_type(16))) __bf16 v16bf;
typedef __attribute__((ext_vector_type(8)))  __bf16 v8bf;
typedef __attribute__((ext_vector_type(8)))  float  v8f;

constexpr int kB  = 8;
constexpr int kC  = 512;
constexpr int kCQ = 64;
constexpr int kW  = 2048;

#define WMMA_BF16(a, b, c) \
  __builtin_amdgcn_wmma_f32_16x16x32_bf16(false, (a), false, (b), (short)0, (c), false, false)

// A-operand (16x32 bf16): lane holds K = {16*half + 0..15} ->
// two contiguous 8-element (16B) groups.
__device__ __forceinline__ v16bf make_a16(const __bf16* g0, const __bf16* g1) {
  v8bf a0 = *(const v8bf*)g0;
  v8bf a1 = *(const v8bf*)g1;
  v16bf r;
#pragma unroll
  for (int e = 0; e < 8; ++e) { r[e] = a0[e]; r[e + 8] = a1[e]; }
  return r;
}

// bf16x3: C += Ah*Bh + Ah*Bl + Al*Bh  (f32 accumulate)
__device__ __forceinline__ v8f bf16x3_acc(v16bf ah, v16bf al, v16bf bh, v16bf bl, v8f c) {
  c = WMMA_BF16(al, bh, c);
  c = WMMA_BF16(ah, bl, c);
  c = WMMA_BF16(ah, bh, c);
  return c;
}

__device__ __forceinline__ void split_f32(float f, __bf16& h, __bf16& l) {
  h = (__bf16)f;
  l = (__bf16)(f - (float)h);
}

// ---------------------------------------------------------------------------
// Elementwise hi/lo split (weights).
// ---------------------------------------------------------------------------
__global__ __launch_bounds__(256) void split_bf16_kernel(
    const float* __restrict__ in, __bf16* __restrict__ hi, __bf16* __restrict__ lo, int n) {
  int i = blockIdx.x * 256 + threadIdx.x;
  if (i < n) {
    __bf16 h, l;
    split_f32(in[i], h, l);
    hi[i] = h; lo[i] = l;
  }
}

// ---------------------------------------------------------------------------
// x[b][c][w] f32  ->  xt[b][w][c] hi/lo bf16 (LDS-tiled 64x64 transpose).
// ---------------------------------------------------------------------------
__global__ __launch_bounds__(256) void transpose_split_kernel(
    const float* __restrict__ X, __bf16* __restrict__ Th, __bf16* __restrict__ Tl) {
  __shared__ __attribute__((aligned(32))) __bf16 sh[64][65];
  __shared__ __attribute__((aligned(32))) __bf16 sl[64][65];
  const int b  = blockIdx.z;
  const int c0 = blockIdx.y * 64;
  const int w0 = blockIdx.x * 64;
  const float* Xb = X + (size_t)b * kC * kW;
  __bf16* Thb = Th + (size_t)b * kW * kC;
  __bf16* Tlb = Tl + (size_t)b * kW * kC;
#pragma unroll
  for (int rep = 0; rep < 16; ++rep) {
    int e = rep * 256 + threadIdx.x;
    int cc = e >> 6, ww = e & 63;
    float f = Xb[(size_t)(c0 + cc) * kW + w0 + ww];
    __bf16 h, l;
    split_f32(f, h, l);
    sh[cc][ww] = h; sl[cc][ww] = l;
  }
  __syncthreads();
#pragma unroll
  for (int rep = 0; rep < 16; ++rep) {
    int e = rep * 256 + threadIdx.x;
    int ww = e >> 6, cc = e & 63;
    Thb[(size_t)(w0 + ww) * kC + c0 + cc] = sh[cc][ww];
    Tlb[(size_t)(w0 + ww) * kC + c0 + cc] = sl[cc][ww];
  }
}

// ---------------------------------------------------------------------------
// Projection GEMM (bf16x3): Y[b][m][w] = sum_c Wm[m][c]*x[b][c][w] + bias[m]
// A = Wm hi/lo [M][C] (row-major), B = xt hi/lo [W][C] (transposed x).
// One wave -> 16(M) x 64(N); K stepped by 32. grid (W/512, M/16, B), 256 thr.
// transposedOut: q/k write [w][M] (S-GEMM operand layout); v writes [M][w].
// ---------------------------------------------------------------------------
__global__ __launch_bounds__(256) void proj_bf16x3_kernel(
    const __bf16* __restrict__ XTh, const __bf16* __restrict__ XTl,
    const __bf16* __restrict__ Wh,  const __bf16* __restrict__ Wl,
    const float* __restrict__ bias,
    __bf16* __restrict__ Oh, __bf16* __restrict__ Ol,
    int M, int transposedOut) {
  const int lane = threadIdx.x & 31;
  const int wave = threadIdx.x >> 5;
  const int half = lane >> 4;
  const int col  = lane & 15;
  const int b    = blockIdx.z;
  const int m0   = blockIdx.y * 16;
  const int n0   = blockIdx.x * 512 + wave * 64;

  const __bf16* Xh = XTh + (size_t)b * kW * kC;
  const __bf16* Xl = XTl + (size_t)b * kW * kC;

  v8f acc[4];
#pragma unroll
  for (int r = 0; r < 8; ++r) {
    float bvv = bias[m0 + r + 8 * half];
    acc[0][r] = bvv; acc[1][r] = bvv; acc[2][r] = bvv; acc[3][r] = bvv;
  }

  for (int k0 = 0; k0 < kC; k0 += 32) {
    const __bf16* arh = Wh + (size_t)(m0 + col) * kC + k0;
    const __bf16* arl = Wl + (size_t)(m0 + col) * kC + k0;
    v16bf ah = make_a16(arh + 16 * half, arh + 16 * half + 8);
    v16bf al = make_a16(arl + 16 * half, arl + 16 * half + 8);
#pragma unroll
    for (int t = 0; t < 4; ++t) {
      const __bf16* bph = Xh + (size_t)(n0 + t * 16 + col) * kC + k0 + 16 * half;
      const __bf16* bpl = Xl + (size_t)(n0 + t * 16 + col) * kC + k0 + 16 * half;
      v16bf bh = *(const v16bf*)bph;
      v16bf bl = *(const v16bf*)bpl;
      acc[t] = bf16x3_acc(ah, al, bh, bl, acc[t]);
    }
  }

  if (transposedOut) {
    __bf16* Ohb = Oh + (size_t)b * kW * M;  // [W][M]
    __bf16* Olb = Ol + (size_t)b * kW * M;
#pragma unroll
    for (int t = 0; t < 4; ++t)
#pragma unroll
      for (int r = 0; r < 8; ++r) {
        int m = m0 + r + 8 * half, w = n0 + t * 16 + col;
        __bf16 h, l;
        split_f32(acc[t][r], h, l);
        Ohb[(size_t)w * M + m] = h;
        Olb[(size_t)w * M + m] = l;
      }
  } else {
    __bf16* Ohb = Oh + (size_t)b * M * kW;  // [M][W]
    __bf16* Olb = Ol + (size_t)b * M * kW;
#pragma unroll
    for (int t = 0; t < 4; ++t)
#pragma unroll
      for (int r = 0; r < 8; ++r) {
        int m = m0 + r + 8 * half, w = n0 + t * 16 + col;
        __bf16 h, l;
        split_f32(acc[t][r], h, l);
        Ohb[(size_t)m * kW + w] = h;
        Olb[(size_t)m * kW + w] = l;
      }
  }
}

// ---------------------------------------------------------------------------
// Attention: workgroup (8 waves) per (batch, i-tile of 32 rows).
// Pass 1: S = q_i^T k (bf16x3) for both 16-row i-subtiles (K loads shared),
// per-row max. Pass 2: per 128-wide j-block each wave stages exp(S-max) hi/lo
// into LDS [i 32][j 128]; all waves accumulate out[c-tile][i] += V * P^T with
// each V load reused across both i-subtiles. Then /rowsum, gamma, +x.
// ---------------------------------------------------------------------------
__global__ __launch_bounds__(256) void attention_kernel(
    const __bf16* __restrict__ QTh, const __bf16* __restrict__ QTl,
    const __bf16* __restrict__ KTh, const __bf16* __restrict__ KTl,
    const __bf16* __restrict__ Vh,  const __bf16* __restrict__ Vl,
    const float* __restrict__ X, const float* __restrict__ gamma,
    float* __restrict__ Out) {
  __shared__ __attribute__((aligned(32))) __bf16 lds_qh[32 * 64];   // [i][qc]
  __shared__ __attribute__((aligned(32))) __bf16 lds_ql[32 * 64];
  __shared__ __attribute__((aligned(32))) __bf16 lds_ph[32 * 128];  // [i][j]
  __shared__ __attribute__((aligned(32))) __bf16 lds_pl[32 * 128];
  __shared__ float lds_part[8 * 32];
  __shared__ float lds_max[32];
  __shared__ float lds_sum[32];

  const int lane = threadIdx.x & 31;
  const int wave = threadIdx.x >> 5;
  const int half = lane >> 4;
  const int col  = lane & 15;
  const int b    = blockIdx.y;
  const int i0   = blockIdx.x * 32;

  const __bf16* Kh  = KTh + (size_t)b * kW * kCQ;  // [w][qc]
  const __bf16* Kl  = KTl + (size_t)b * kW * kCQ;
  const __bf16* Vhb = Vh + (size_t)b * kC * kW;    // [c][w]
  const __bf16* Vlb = Vl + (size_t)b * kC * kW;
  const __bf16* Qhb = QTh + (size_t)b * kW * kCQ;
  const __bf16* Qlb = QTl + (size_t)b * kW * kCQ;

  // Stage q tile (32 rows x 64 qc, hi/lo).
#pragma unroll
  for (int rep = 0; rep < 8; ++rep) {
    int e = rep * 256 + threadIdx.x;
    int ii = e >> 6, qc = e & 63;
    lds_qh[e] = Qhb[(size_t)(i0 + ii) * kCQ + qc];
    lds_ql[e] = Qlb[(size_t)(i0 + ii) * kCQ + qc];
  }
  __syncthreads();

  // ---------------- pass 1: row maxima ----------------
  float rmax[2][8];
#pragma unroll
  for (int u = 0; u < 2; ++u)
#pragma unroll
    for (int r = 0; r < 8; ++r) rmax[u][r] = -3.4e38f;

  for (int jt = wave; jt < 128; jt += 8) {
    const int j0 = jt * 16;
    v8f s[2] = {};
#pragma unroll
    for (int ch = 0; ch < 2; ++ch) {
      const int k0 = ch * 32;
      const __bf16* kbh = Kh + (size_t)(j0 + col) * kCQ + k0 + 16 * half;
      const __bf16* kbl = Kl + (size_t)(j0 + col) * kCQ + k0 + 16 * half;
      v16bf bh = *(const v16bf*)kbh;   // shared across both i-subtiles
      v16bf bl = *(const v16bf*)kbl;
#pragma unroll
      for (int u = 0; u < 2; ++u) {
        const __bf16* qa = lds_qh + (u * 16 + col) * 64 + k0;
        const __bf16* ql = lds_ql + (u * 16 + col) * 64 + k0;
        v16bf ah = make_a16(qa + 16 * half, qa + 16 * half + 8);
        v16bf al = make_a16(ql + 16 * half, ql + 16 * half + 8);
        s[u] = bf16x3_acc(ah, al, bh, bl, s[u]);
      }
    }
#pragma unroll
    for (int u = 0; u < 2; ++u)
#pragma unroll
      for (int r = 0; r < 8; ++r) {
        float m = s[u][r];
        m = fmaxf(m, __shfl_xor(m, 1, 32));
        m = fmaxf(m, __shfl_xor(m, 2, 32));
        m = fmaxf(m, __shfl_xor(m, 4, 32));
        m = fmaxf(m, __shfl_xor(m, 8, 32));
        rmax[u][r] = fmaxf(rmax[u][r], m);
      }
  }
  if (lane == 0 || lane == 16) {
#pragma unroll
    for (int u = 0; u < 2; ++u)
#pragma unroll
      for (int r = 0; r < 8; ++r)
        lds_part[wave * 32 + u * 16 + 8 * half + r] = rmax[u][r];
  }
  __syncthreads();
  if (threadIdx.x < 32) {
    float m = -3.4e38f;
    for (int w2 = 0; w2 < 8; ++w2) m = fmaxf(m, lds_part[w2 * 32 + threadIdx.x]);
    lds_max[threadIdx.x] = m;
  }
  __syncthreads();

  float mx[2][8];
#pragma unroll
  for (int u = 0; u < 2; ++u)
#pragma unroll
    for (int r = 0; r < 8; ++r) mx[u][r] = lds_max[u * 16 + r + 8 * half];

  // ---------------- pass 2: exp + output accumulation ----------------
  v8f acc[4][2] = {};
  float ssum[2][8];
#pragma unroll
  for (int u = 0; u < 2; ++u)
#pragma unroll
    for (int r = 0; r < 8; ++r) ssum[u][r] = 0.0f;

  const int ct0 = wave * 4;

  for (int jb = 0; jb < 16; ++jb) {
    // Stage this wave's 16-wide j-tile of exp(S - max), both i-subtiles.
    const int j0 = jb * 128 + wave * 16;
    v8f s[2] = {};
#pragma unroll
    for (int ch = 0; ch < 2; ++ch) {
      const int k0 = ch * 32;
      const __bf16* kbh = Kh + (size_t)(j0 + col) * kCQ + k0 + 16 * half;
      const __bf16* kbl = Kl + (size_t)(j0 + col) * kCQ + k0 + 16 * half;
      v16bf bh = *(const v16bf*)kbh;
      v16bf bl = *(const v16bf*)kbl;
#pragma unroll
      for (int u = 0; u < 2; ++u) {
        const __bf16* qa = lds_qh + (u * 16 + col) * 64 + k0;
        const __bf16* ql = lds_ql + (u * 16 + col) * 64 + k0;
        v16bf ah = make_a16(qa + 16 * half, qa + 16 * half + 8);
        v16bf al = make_a16(ql + 16 * half, ql + 16 * half + 8);
        s[u] = bf16x3_acc(ah, al, bh, bl, s[u]);
      }
    }
#pragma unroll
    for (int u = 0; u < 2; ++u)
#pragma unroll
      for (int r = 0; r < 8; ++r) {
        float p = __expf(s[u][r] - mx[u][r]);   // row i, col j = col
        __bf16 ph, pl;
        split_f32(p, ph, pl);
        const int ii = u * 16 + r + 8 * half, jl = wave * 16 + col;
        lds_ph[ii * 128 + jl] = ph;
        lds_pl[ii * 128 + jl] = pl;
        float su = (float)ph + (float)pl;  // sum exactly what the GEMM sees
        su += __shfl_xor(su, 1, 32);
        su += __shfl_xor(su, 2, 32);
        su += __shfl_xor(su, 4, 32);
        su += __shfl_xor(su, 8, 32);
        ssum[u][r] += su;
      }
    __syncthreads();

    // out[c-tile 16][i 16] += V[c, j-block] * P^T[j-block, i]; V loads shared
    // across both i-subtiles.
#pragma unroll
    for (int t = 0; t < 4; ++t) {
      const int c0 = (ct0 + t) * 16;
      const __bf16* vh = Vhb + (size_t)(c0 + col) * kW + (size_t)jb * 128;
      const __bf16* vl = Vlb + (size_t)(c0 + col) * kW + (size_t)jb * 128;
#pragma unroll
      for (int ch = 0; ch < 4; ++ch) {
        const int k0 = ch * 32;
        v16bf ah = make_a16(vh + k0 + 16 * half, vh + k0 + 16 * half + 8);
        v16bf al = make_a16(vl + k0 + 16 * half, vl + k0 + 16 * half + 8);
#pragma unroll
        for (int u = 0; u < 2; ++u) {
          v16bf bh = *(const v16bf*)(lds_ph + (u * 16 + col) * 128 + k0 + 16 * half);
          v16bf bl = *(const v16bf*)(lds_pl + (u * 16 + col) * 128 + k0 + 16 * half);
          acc[t][u] = bf16x3_acc(ah, al, bh, bl, acc[t][u]);
        }
      }
    }
    __syncthreads();
  }

  // Row-sum reduction across waves.
  if (lane == 0 || lane == 16) {
#pragma unroll
    for (int u = 0; u < 2; ++u)
#pragma unroll
      for (int r = 0; r < 8; ++r)
        lds_part[wave * 32 + u * 16 + 8 * half + r] = ssum[u][r];
  }
  __syncthreads();
  if (threadIdx.x < 32) {
    float su = 0.0f;
    for (int w2 = 0; w2 < 8; ++w2) su += lds_part[w2 * 32 + threadIdx.x];
    lds_sum[threadIdx.x] = su;
  }
  __syncthreads();

  const float sinv0 = 1.0f / lds_sum[col];
  const float sinv1 = 1.0f / lds_sum[16 + col];
  const float g = gamma[0];
  const float* Xb = X + (size_t)b * kC * kW;
  float*       Ob = Out + (size_t)b * kC * kW;
#pragma unroll
  for (int t = 0; t < 4; ++t) {
    const int c0 = (ct0 + t) * 16;
#pragma unroll
    for (int u = 0; u < 2; ++u) {
      const float sinv = u ? sinv1 : sinv0;
#pragma unroll
      for (int r = 0; r < 8; ++r) {
        const size_t idx = (size_t)(c0 + r + 8 * half) * kW + i0 + u * 16 + col;
        Ob[idx] = g * (acc[t][u][r] * sinv) + Xb[idx];
      }
    }
  }
}

// ---------------------------------------------------------------------------
extern "C" void kernel_launch(void* const* d_in, const int* in_sizes, int n_in,
                              void* d_out, int out_size, void* d_ws, size_t ws_size,
                              hipStream_t stream) {
  const float* x     = (const float*)d_in[0];
  const float* Wq    = (const float*)d_in[1];
  const float* bq    = (const float*)d_in[2];
  const float* Wk    = (const float*)d_in[3];
  const float* bk    = (const float*)d_in[4];
  const float* Wv    = (const float*)d_in[5];
  const float* bv    = (const float*)d_in[6];
  const float* gamma = (const float*)d_in[7];
  float* out = (float*)d_out;

  const size_t nx  = (size_t)kB * kC * kW;   // 8.39M elems
  const size_t nqt = (size_t)kB * kW * kCQ;  // 1.05M elems
  __bf16* p = (__bf16*)d_ws;
  __bf16* xt_h = p; p += nx;
  __bf16* xt_l = p; p += nx;
  __bf16* v_h  = p; p += nx;
  __bf16* v_l  = p; p += nx;
  __bf16* qt_h = p; p += nqt;
  __bf16* qt_l = p; p += nqt;
  __bf16* kt_h = p; p += nqt;
  __bf16* kt_l = p; p += nqt;
  __bf16* wq_h = p; p += kCQ * kC;
  __bf16* wq_l = p; p += kCQ * kC;
  __bf16* wk_h = p; p += kCQ * kC;
  __bf16* wk_l = p; p += kCQ * kC;
  __bf16* wv_h = p; p += (size_t)kC * kC;
  __bf16* wv_l = p; p += (size_t)kC * kC;

  // hi/lo splits of weights and (transposed) activations
  split_bf16_kernel<<<(kCQ * kC + 255) / 256, 256, 0, stream>>>(Wq, wq_h, wq_l, kCQ * kC);
  split_bf16_kernel<<<(kCQ * kC + 255) / 256, 256, 0, stream>>>(Wk, wk_h, wk_l, kCQ * kC);
  split_bf16_kernel<<<(kC * kC + 255) / 256, 256, 0, stream>>>(Wv, wv_h, wv_l, kC * kC);
  transpose_split_kernel<<<dim3(kW / 64, kC / 64, kB), 256, 0, stream>>>(x, xt_h, xt_l);

  // q = Wq x + bq (transposed out), k = Wk x + bk (transposed), v = Wv x + bv
  proj_bf16x3_kernel<<<dim3(kW / 512, kCQ / 16, kB), 256, 0, stream>>>(
      xt_h, xt_l, wq_h, wq_l, bq, qt_h, qt_l, kCQ, 1);
  proj_bf16x3_kernel<<<dim3(kW / 512, kCQ / 16, kB), 256, 0, stream>>>(
      xt_h, xt_l, wk_h, wk_l, bk, kt_h, kt_l, kCQ, 1);
  proj_bf16x3_kernel<<<dim3(kW / 512, kC / 16, kB), 256, 0, stream>>>(
      xt_h, xt_l, wv_h, wv_l, bv, v_h, v_l, kC, 0);

  // softmax(q^T k) applied to v, + gamma residual
  attention_kernel<<<dim3(kW / 32, kB), 256, 0, stream>>>(
      qt_h, qt_l, kt_h, kt_l, v_h, v_l, x, gamma, out);
}
